// TemporalAttentionLayer_38817914421793
// MI455X (gfx1250) — matrix operand order repaired and checked by reading, not compiled, not run on hardware
//
#include <hip/hip_runtime.h>
#include <hip/hip_bf16.h>
#include <math.h>

#define DEVFN __device__ __forceinline__

typedef __bf16 bf16_t;
typedef __attribute__((ext_vector_type(16))) __bf16 v16bf;
typedef __attribute__((ext_vector_type(8)))  float  v8f;
typedef int v4i __attribute__((ext_vector_type(4)));

// ---------------- problem dimensions ----------------
#define BB   512     // batch
#define RR   5       // relations
#define NN   64      // neighbors
#define FF   172     // node feat
#define TT   100     // time feat
#define EFD  172     // edge feat
#define EE   272     // E = F+T
#define KK   444     // K = F+EF+T
#define KP   448     // K padded to x32
#define EPK  288     // E padded to x32 (as reduction dim)
#define BN   32768   // B*N (rows per relation in KV GEMM)
#define DH   136     // head dim
#define C1P  1536    // merge input padded (1532 -> 1536)
#define HP   192     // hidden padded (172 -> 192)
#define OP   176     // out padded (172 -> 176)
#define OUTD 172

#define BROWS 256        // rows staged per block (8 waves x MT(2) x 16)
#define LPITCH 80        // LDS row pitch (bytes) for one 32-col k-chunk (64B data + pad)

// ---------------- async global->LDS (CDNA5) with fallbacks ----------------
#if defined(__HIP_DEVICE_COMPILE__) && defined(__gfx1250__)
#  if __has_builtin(__builtin_amdgcn_global_load_async_to_lds_b128)
#    define USE_ASYNC_LDS 1
#  endif
#endif

DEVFN void async_copy16(const bf16_t* g, char* l) {
#ifdef USE_ASYNC_LDS
  __builtin_amdgcn_global_load_async_to_lds_b128(
      (__attribute__((address_space(1))) v4i*)(size_t)g,
      (__attribute__((address_space(3))) v4i*)l,
      0, 0);
#else
  *(uint4*)l = *(const uint4*)g;
#endif
}

DEVFN void wait_async() {
#ifdef USE_ASYNC_LDS
#  if __has_builtin(__builtin_amdgcn_s_wait_asynccnt)
  __builtin_amdgcn_s_wait_asynccnt(0);
#  else
  asm volatile("s_wait_asynccnt 0x0" ::: "memory");
#  endif
#endif
}

// ---------------- WMMA fragment loads ----------------
union FragB {
  v16bf v;
  uint4 u[2];
};

// A fragment from an LDS k-chunk (rows x 32 cols, pitch LPITCH bytes).
// lane l: row = base+(l&15); half = l>>4
// VGPR0..7 : K = 16*half + 0..15
DEVFN v16bf load_frag_a_lds(const char* buf, int row, int half) {
  FragB f;
  const char* p = buf + row * LPITCH + 32 * half;
  f.u[0] = *(const uint4*)p;
  f.u[1] = *(const uint4*)(p + 16);
  return f.v;
}

// B fragment: 32x16 bf16 from row-major weights W[n][k], lane l: col n = n0+(l&15);
// VGPR0..7 packed K = k0 + 16*half + 0..15 (two contiguous 16B chunks)
DEVFN v16bf load_frag_b(const bf16_t* __restrict__ Bw, int col, int ldb, int k0, int half) {
  FragB f;
  const char* p = (const char*)Bw + (((size_t)col * ldb) + (size_t)(k0 + 16 * half)) * 2;
  f.u[0] = *(const uint4*)p;
  f.u[1] = *(const uint4*)(p + 16);
  return f.v;
}

// Stage one 256-row x 32-col A chunk into LDS (block-wide, 4 x b128 per thread).
DEVFN void stage_chunk(const bf16_t* __restrict__ A, int lda, int k0, char* buf, int tid) {
#pragma unroll
  for (int i = 0; i < 4; ++i) {
    const int id = tid + 256 * i;       // 0..1023
    const int row = id >> 2;            // 0..255
    const int c = id & 3;               // 16B chunk within 64B row
    async_copy16(A + (size_t)row * lda + (size_t)(k0 + c * 8), buf + row * LPITCH + c * 16);
  }
}

// ---------------- generic WMMA GEMM ----------------
// C[m][n] = sum_k A[m][k] * W[n][k] + bias[n]  (optional 2nd weight sharing A; MT=2)
template<int NB, int NT, bool RELU, bool OUT_BF16, bool INVZERO>
DEVFN void compute_group(const bf16_t* __restrict__ A, int lda,
                         const bf16_t* __restrict__ B0, const bf16_t* __restrict__ B1, int ldb,
                         const float* __restrict__ bias0, const float* __restrict__ bias1,
                         void* C0, void* C1, int ldcRow, int colBase,
                         int mBlock, int n0, int Kred, int colLimit,
                         const int* __restrict__ invP, int invStride,
                         char* sm0, char* sm1, int tid)
{
  constexpr int IB1 = (NB > 1) ? 1 : 0;   // clamped index (dead for NB==1)
  const int lane = tid & 31;
  const int wave = tid >> 5;
  const int lnn  = lane & 15;
  const int half = lane >> 4;

  v8f zero = {};
  v8f acc[NB][2][NT];
#pragma unroll
  for (int b = 0; b < NB; ++b)
#pragma unroll
    for (int mt = 0; mt < 2; ++mt)
#pragma unroll
      for (int t = 0; t < NT; ++t) acc[b][mt][t] = zero;

  const int nsteps = Kred / 32;
  stage_chunk(A, lda, 0, sm0, tid);
  for (int s = 0; s < nsteps; ++s) {
    wait_async();              // own async copies (current buffer) complete
    __syncthreads();           // all waves loaded current buf & done reading next-buf's old data
    char* cur = (s & 1) ? sm1 : sm0;
    char* nxt = (s & 1) ? sm0 : sm1;
    if (s + 1 < nsteps) stage_chunk(A, lda, (s + 1) * 32, nxt, tid);

    v16bf afr[2];
#pragma unroll
    for (int mt = 0; mt < 2; ++mt)
      afr[mt] = load_frag_a_lds(cur, wave * 32 + mt * 16 + lnn, half);
#pragma unroll
    for (int t = 0; t < NT; ++t) {
      const int col = n0 + 16 * t + lnn;
      v16bf b0 = load_frag_b(B0, col, ldb, s * 32, half);
#pragma unroll
      for (int mt = 0; mt < 2; ++mt)
        acc[0][mt][t] = __builtin_amdgcn_wmma_f32_16x16x32_bf16(
            false, afr[mt], false, b0, (short)0, acc[0][mt][t], false, false);
      if (NB > 1) {
        v16bf b1f = load_frag_b(B1, col, ldb, s * 32, half);
#pragma unroll
        for (int mt = 0; mt < 2; ++mt)
          acc[IB1][mt][t] = __builtin_amdgcn_wmma_f32_16x16x32_bf16(
              false, afr[mt], false, b1f, (short)0, acc[IB1][mt][t], false, false);
      }
    }
  }
  __syncthreads();   // protect LDS against the next group's prologue staging

  // epilogue
#pragma unroll
  for (int t = 0; t < NT; ++t) {
    const int col = n0 + 16 * t + lnn;
    const float bv0 = bias0[col];
    const float bv1 = (NB > 1) ? bias1[col] : 0.f;
#pragma unroll
    for (int mt = 0; mt < 2; ++mt) {
#pragma unroll
      for (int v = 0; v < 8; ++v) {
        const int rowI = mBlock + wave * 32 + mt * 16 + v + 8 * half;
        const size_t oi = (size_t)rowI * ldcRow + (size_t)colBase + col;
        float z0 = acc[0][mt][t][v] + bv0;
        if (RELU) z0 = fmaxf(z0, 0.f);
        if (INVZERO && invP[(size_t)rowI * invStride]) z0 = 0.f;
        if (col < colLimit) {
          if (OUT_BF16) ((bf16_t*)C0)[oi] = (bf16_t)z0;
          else          ((float*)C0)[oi] = z0;
        }
        if (NB > 1) {
          float z1 = acc[IB1][mt][t][v] + bv1;
          if (RELU) z1 = fmaxf(z1, 0.f);
          if (col < colLimit) {
            if (OUT_BF16) ((bf16_t*)C1)[oi] = (bf16_t)z1;
            else          ((float*)C1)[oi] = z1;
          }
        }
      }
    }
  }
}

template<int NB, int NT, bool RELU, bool OUT_BF16, bool INVZERO>
__global__ __launch_bounds__(256) void gemm_wmma_kernel(
    const bf16_t* __restrict__ A, size_t aStride, int lda,
    const bf16_t* __restrict__ B0, const bf16_t* __restrict__ B1, size_t bStride, int ldb,
    const float* __restrict__ bias0, const float* __restrict__ bias1, int biasStride,
    void* __restrict__ C0v, void* __restrict__ C1v, size_t cStride, int ldcRow, int cColStride,
    int Kred, int Nout, int colLimit,
    const int* __restrict__ invBase, int invStride)
{
  __shared__ char smem[2][BROWS * LPITCH];   // 2 x 20KB double buffer
  const int r = blockIdx.y;
  A  += (size_t)r * aStride;
  B0 += (size_t)r * bStride;
  const bf16_t* B1p = (NB > 1) ? (B1 + (size_t)r * bStride) : nullptr;
  bias0 += (size_t)r * biasStride;
  const float* bias1p = (NB > 1) ? (bias1 + (size_t)r * biasStride) : nullptr;
  const int esz = OUT_BF16 ? 2 : 4;
  void* C0 = (void*)((char*)C0v + (size_t)r * cStride * esz);
  void* C1 = (NB > 1) ? (void*)((char*)C1v + (size_t)r * cStride * esz) : nullptr;
  const int* invP = INVZERO ? (invBase + r) : nullptr;
  const int colBase = r * cColStride;

  const int tid = threadIdx.x;
  const int mBlock = blockIdx.x * BROWS;
  const bf16_t* Ablk = A + (size_t)mBlock * lda;

  int n0 = 0;
  for (; n0 + 16 * NT <= Nout; n0 += 16 * NT)
    compute_group<NB, NT, RELU, OUT_BF16, INVZERO>(Ablk, lda, B0, B1p, ldb, bias0, bias1p,
        C0, C1, ldcRow, colBase, mBlock, n0, Kred, colLimit, invP, invStride,
        smem[0], smem[1], tid);
  for (; n0 < Nout; n0 += 16)
    compute_group<NB, 1, RELU, OUT_BF16, INVZERO>(Ablk, lda, B0, B1p, ldb, bias0, bias1p,
        C0, C1, ldcRow, colBase, mBlock, n0, Kred, colLimit, invP, invStride,
        smem[0], smem[1], tid);
}

// ---------------- packing kernels (fp32 -> padded bf16) ----------------
__global__ __launch_bounds__(256) void pack_kin_kernel(
    const float* __restrict__ nf, const float* __restrict__ ef,
    const float* __restrict__ tf, bf16_t* __restrict__ kin)
{
  const int idx = blockIdx.x * blockDim.x + threadIdx.x; // one thread per 8 elems
  const int total = RR * BN * (KP / 8);
  if (idx >= total) return;
  const int chunk = idx % (KP / 8);
  const int row   = idx / (KP / 8);          // r*BN + b*NN + n
  const int r = row / BN;
  const int m = row % BN;
  const int bI = m / NN, n = m % NN;
  const size_t sb = ((size_t)(bI * RR + r) * NN + n);
  const int k0 = chunk * 8;
  alignas(16) bf16_t ov[8];
#pragma unroll
  for (int i = 0; i < 8; ++i) {
    const int k = k0 + i;
    float v;
    if      (k < FF)        v = nf[sb * FF  + k];
    else if (k < FF + EFD)  v = ef[sb * EFD + (k - FF)];
    else if (k < KK)        v = tf[sb * TT  + (k - FF - EFD)];
    else                    v = 0.f;
    ov[i] = (bf16_t)v;
  }
  *(uint4*)(kin + (size_t)row * KP + k0) = *(const uint4*)ov;
}

__global__ __launch_bounds__(256) void pack_x_kernel(
    const float* __restrict__ nodef, const float* __restrict__ timef, bf16_t* __restrict__ xq)
{
  const int idx = blockIdx.x * blockDim.x + threadIdx.x;
  if (idx >= BB * EPK) return;
  const int bI = idx / EPK, e = idx % EPK;
  float v = 0.f;
  if (e < FF)      v = nodef[bI * FF + e];
  else if (e < EE) v = timef[bI * TT + (e - FF)];
  xq[idx] = (bf16_t)v;
}

__global__ __launch_bounds__(256) void pack_pad_bf16_kernel(
    const float* __restrict__ src, bf16_t* __restrict__ dst,
    int rowsD, int colsD, int rowsS, int colsS)
{
  const int idx = blockIdx.x * blockDim.x + threadIdx.x;
  if (idx >= rowsD * colsD) return;
  const int rI = idx / colsD, c = idx % colsD;
  const float v = (rI < rowsS && c < colsS) ? src[(size_t)rI * colsS + c] : 0.f;
  dst[idx] = (bf16_t)v;
}

__global__ __launch_bounds__(256) void pad_f32_kernel(
    const float* __restrict__ src, float* __restrict__ dst, int nD, int nS)
{
  const int idx = blockIdx.x * blockDim.x + threadIdx.x;
  if (idx >= nD) return;
  dst[idx] = (idx < nS) ? src[idx] : 0.f;
}

__global__ __launch_bounds__(256) void pack_merge_src_kernel(
    const float* __restrict__ src, bf16_t* __restrict__ mrg)
{
  const int idx = blockIdx.x * blockDim.x + threadIdx.x;
  if (idx >= BB * (C1P - RR * EE)) return;
  const int cspan = C1P - RR * EE;          // 176
  const int bI = idx / cspan, c = idx % cspan;
  const float v = (c < FF) ? src[bI * FF + c] : 0.f;
  mrg[(size_t)bI * C1P + (RR * EE) + c] = (bf16_t)v;
}

// ---------------- attention (batched M=1 GEMV, VALU) ----------------
__global__ __launch_bounds__(128) void attn_kernel(
    const float* __restrict__ qbuf, const bf16_t* __restrict__ kbuf,
    const bf16_t* __restrict__ vbuf, const int* __restrict__ maskp,
    bf16_t* __restrict__ attnb, int* __restrict__ invb, float* __restrict__ wavg)
{
  const int bI = blockIdx.x;
  const int r  = blockIdx.y;
  __shared__ float qs[EE];
  __shared__ float sc[2][NN];
  __shared__ float wsf[2][NN];
  __shared__ int   ms[NN];
  __shared__ int   sInv;
  __shared__ float mx[2], sm[2];
  const int tid = threadIdx.x;

  if (tid < NN) ms[tid] = maskp[((size_t)r * BB + bI) * NN + tid];
  for (int e = tid; e < EE; e += 128)
    qs[e] = qbuf[((size_t)bI * RR + r) * EE + e];
  __syncthreads();
  if (tid == 0) {
    int inv = 1;
    for (int n = 0; n < NN; ++n) inv &= (ms[n] != 0);
    sInv = inv;
  }
  __syncthreads();
  const int inv = sInv;
  const size_t kvbase = ((size_t)r * BN + (size_t)bI * NN) * EE;

  if (tid < NN) {
    const int n = tid;
    const int m2 = (ms[n] != 0) && !(inv && n == 0);  // un-pad neighbor 0 if all padded
    const bf16_t* krow = kbuf + kvbase + (size_t)n * EE;
    float a0 = 0.f, a1 = 0.f;
    for (int d = 0; d < DH; ++d) {
      a0 += qs[d]      * (float)krow[d];
      a1 += qs[DH + d] * (float)krow[DH + d];
    }
    const float scale = 0.08574929257125441f;   // 1/sqrt(136)
    const float ninf = -__builtin_inff();
    sc[0][n] = m2 ? ninf : a0 * scale;
    sc[1][n] = m2 ? ninf : a1 * scale;
  }
  __syncthreads();
  if (tid < 2) {
    float m = -__builtin_inff();
    for (int n = 0; n < NN; ++n) m = fmaxf(m, sc[tid][n]);
    float s = 0.f;
    for (int n = 0; n < NN; ++n) s += __expf(sc[tid][n] - m);
    mx[tid] = m; sm[tid] = s;
  }
  __syncthreads();
  if (tid < NN) {
    const int n = tid;
    const float w0 = __expf(sc[0][n] - mx[0]) / sm[0];
    const float w1 = __expf(sc[1][n] - mx[1]) / sm[1];
    wsf[0][n] = w0; wsf[1][n] = w1;
    wavg[((size_t)bI * RR + r) * NN + n] = inv ? 0.f : 0.5f * (w0 + w1);
  }
  __syncthreads();
  for (int e = tid; e < EPK; e += 128) {
    float acc = 0.f;
    if (e < EE) {
      const float* wr = wsf[(e >= DH) ? 1 : 0];
      const bf16_t* vcol = vbuf + kvbase + e;
      for (int n = 0; n < NN; ++n) acc += wr[n] * (float)vcol[(size_t)n * EE];
    }
    attnb[((size_t)bI * RR + r) * EPK + e] = (bf16_t)acc;  // padded cols -> 0
  }
  if (tid == 0) invb[bI * RR + r] = inv;
}

// ---------------- host launcher ----------------
static inline unsigned cdivu(size_t a, unsigned b) { return (unsigned)((a + b - 1) / b); }

extern "C" void kernel_launch(void* const* d_in, const int* in_sizes, int n_in,
                              void* d_out, int out_size, void* d_ws, size_t ws_size,
                              hipStream_t stream) {
  const float* src_node = (const float*)d_in[0];
  const float* src_time = (const float*)d_in[1];
  const float* neigh_f  = (const float*)d_in[2];
  const float* neigh_t  = (const float*)d_in[3];
  const float* edge_f   = (const float*)d_in[4];
  const int*   maskp    = (const int*)d_in[5];
  const float* Wq = (const float*)d_in[6];
  const float* Wk = (const float*)d_in[7];
  const float* Wv = (const float*)d_in[8];
  const float* bq = (const float*)d_in[9];
  const float* bk = (const float*)d_in[10];
  const float* bv = (const float*)d_in[11];
  const float* Wo = (const float*)d_in[12];
  const float* bo = (const float*)d_in[13];
  const float* W1 = (const float*)d_in[14];
  const float* b1 = (const float*)d_in[15];
  const float* W2 = (const float*)d_in[16];
  const float* b2 = (const float*)d_in[17];
  (void)in_sizes; (void)n_in; (void)out_size; (void)ws_size;

  char* ws = (char*)d_ws;
  size_t off = 0;
  auto take = [&](size_t bytes) -> char* {
    char* p = ws + off;
    off += (bytes + 255) & ~(size_t)255;
    return p;
  };
  bf16_t* kin  = (bf16_t*)take((size_t)RR * BN * KP * 2);    // ~140 MB
  bf16_t* xq   = (bf16_t*)take((size_t)BB * EPK * 2);
  bf16_t* wqP  = (bf16_t*)take((size_t)RR * EE * EPK * 2);
  bf16_t* wkP  = (bf16_t*)take((size_t)RR * EE * KP * 2);
  bf16_t* wvP  = (bf16_t*)take((size_t)RR * EE * KP * 2);
  bf16_t* woP  = (bf16_t*)take((size_t)RR * EE * EPK * 2);
  bf16_t* w1P  = (bf16_t*)take((size_t)HP * C1P * 2);
  bf16_t* w2P  = (bf16_t*)take((size_t)OP * HP * 2);
  float*  b1P  = (float*)take((size_t)HP * 4);
  float*  b2P  = (float*)take((size_t)OP * 4);
  float*  qb   = (float*)take((size_t)BB * RR * EE * 4);
  bf16_t* kb   = (bf16_t*)take((size_t)RR * BN * EE * 2);    // ~85 MB
  bf16_t* vb   = (bf16_t*)take((size_t)RR * BN * EE * 2);    // ~85 MB
  bf16_t* attnb= (bf16_t*)take((size_t)BB * RR * EPK * 2);
  bf16_t* mrg  = (bf16_t*)take((size_t)BB * C1P * 2);
  bf16_t* hb   = (bf16_t*)take((size_t)BB * HP * 2);
  int*    invb = (int*)take((size_t)BB * RR * 4);

  // ---- pack activations & weights to padded bf16 ----
  pack_kin_kernel<<<cdivu((size_t)RR * BN * (KP / 8), 256), 256, 0, stream>>>(neigh_f, edge_f, neigh_t, kin);
  pack_x_kernel<<<cdivu((size_t)BB * EPK, 256), 256, 0, stream>>>(src_node, src_time, xq);
  pack_pad_bf16_kernel<<<cdivu((size_t)RR * EE * EPK, 256), 256, 0, stream>>>(Wq, wqP, RR * EE, EPK, RR * EE, EE);
  pack_pad_bf16_kernel<<<cdivu((size_t)RR * EE * KP, 256), 256, 0, stream>>>(Wk, wkP, RR * EE, KP, RR * EE, KK);
  pack_pad_bf16_kernel<<<cdivu((size_t)RR * EE * KP, 256), 256, 0, stream>>>(Wv, wvP, RR * EE, KP, RR * EE, KK);
  pack_pad_bf16_kernel<<<cdivu((size_t)RR * EE * EPK, 256), 256, 0, stream>>>(Wo, woP, RR * EE, EPK, RR * EE, EE);
  pack_pad_bf16_kernel<<<cdivu((size_t)HP * C1P, 256), 256, 0, stream>>>(W1, w1P, HP, C1P, OUTD, RR * EE + FF);
  pack_pad_bf16_kernel<<<cdivu((size_t)OP * HP, 256), 256, 0, stream>>>(W2, w2P, OP, HP, OUTD, FF);
  pad_f32_kernel<<<1, 256, 0, stream>>>(b1, b1P, HP, FF);
  pad_f32_kernel<<<1, 256, 0, stream>>>(b2, b2P, OP, OUTD);

  // ---- q = x @ Wq^T + bq  (per relation) -> qb[b][r][272] (f32) ----
  {
    dim3 g(BB / BROWS, RR);
    gemm_wmma_kernel<1, 4, false, false, false><<<g, 256, 0, stream>>>(
        xq, 0, EPK,
        wqP, nullptr, (size_t)EE * EPK, EPK,
        bq, nullptr, EE,
        qb, nullptr, 0, RR * EE, EE,
        EPK, EE, EE,
        nullptr, 0);
  }
  // ---- fused K/V projections: kin @ {Wk,Wv}^T + {bk,bv} -> kb/vb bf16 ----
  {
    dim3 g(BN / BROWS, RR);
    gemm_wmma_kernel<2, 2, false, true, false><<<g, 256, 0, stream>>>(
        kin, (size_t)BN * KP, KP,
        wkP, wvP, (size_t)EE * KP, KP,
        bk, bv, EE,
        kb, vb, (size_t)BN * EE, EE, 0,
        KP, EE, EE,
        nullptr, 0);
  }
  // ---- attention: scores, mask, softmax, weighted sum, w_avg ----
  {
    dim3 g(BB, RR);
    attn_kernel<<<g, 128, 0, stream>>>(qb, kb, vb, maskp, attnb, invb,
                                       (float*)d_out + (size_t)BB * OUTD);
  }
  // ---- out = attn @ Wo^T + bo, zero where invalid, write into merge input ----
  {
    dim3 g(BB / BROWS, RR);
    gemm_wmma_kernel<1, 4, false, true, true><<<g, 256, 0, stream>>>(
        attnb, EPK, RR * EPK,
        woP, nullptr, (size_t)EE * EPK, EPK,
        bo, nullptr, EE,
        mrg, nullptr, 0, C1P, EE,
        EPK, EE, EE,
        invb, RR);
  }
  pack_merge_src_kernel<<<cdivu((size_t)BB * (C1P - RR * EE), 256), 256, 0, stream>>>(src_node, mrg);

  // ---- h = relu(merge_in @ W1^T + b1) -> hb bf16 [B][192] ----
  {
    dim3 g(BB / BROWS, 1);
    gemm_wmma_kernel<1, 4, true, true, false><<<g, 256, 0, stream>>>(
        mrg, 0, C1P,
        w1P, nullptr, 0, C1P,
        b1P, nullptr, 0,
        hb, nullptr, 0, HP, 0,
        C1P, HP, HP,
        nullptr, 0);
  }
  // ---- merged = h @ W2^T + b2 -> d_out f32 [B][172] ----
  {
    dim3 g(BB / BROWS, 1);
    gemm_wmma_kernel<1, 4, false, false, false><<<g, 256, 0, stream>>>(
        hb, 0, HP,
        w2P, nullptr, 0, HP,
        b2P, nullptr, 0,
        d_out, nullptr, 0, OUTD, 0,
        HP, OP, OUTD,
        nullptr, 0);
  }
}
